// SubGraph_45535243272609
// MI455X (gfx1250) — compile-verified, matrix operand order checked by reading, not run on hardware
//
#include <hip/hip_runtime.h>

// bf16 scalar + WMMA vector types (CDNA5 gfx1250, wave32)
typedef __bf16 bhalf;
typedef __bf16 v16bf __attribute__((ext_vector_type(16)));
typedef float  v8f   __attribute__((ext_vector_type(8)));

// ---------------------------------------------------------------------------
// Weight fragment packing: W [Kdim,64] fp32 (row-major, as in the reference:
// x @ W) -> bf16 fragments laid out per the CDNA5 WMMA 16-bit B layout
// (16x16x32 tiles): frag[ks][nt][lane][h], lane<16: N=lane, K = {0..7,16..23};
// lane>=16: N=lane-16, K = {8..15,24..31} (+ks*32). Rows beyond the true K
// (e.g. d=8/10 padded to 32) are zero.
// ---------------------------------------------------------------------------
__global__ void pack_frag(const float* __restrict__ W, bhalf* __restrict__ dst,
                          int srcK, int total) {
  int idx = blockIdx.x * 256 + threadIdx.x;
  if (idx >= total) return;
  int ks   = idx >> 11;          // 2048 elems per K-step (4 ntiles * 32 lanes * 16)
  int rem  = idx & 2047;
  int nt   = rem >> 9;
  int lane = (rem >> 4) & 31;
  int h    = rem & 15;
  int koff = (lane >> 4) << 3;
  int k    = ks * 32 + koff + h + ((h & 8) ? 8 : 0);
  int col  = nt * 16 + (lane & 15);
  float v  = (k < srcK) ? W[k * 64 + col] : 0.0f;
  dst[idx] = (bhalf)v;
}

// ---------------------------------------------------------------------------
// Per-branch fused kernel
// ---------------------------------------------------------------------------
struct BranchArgs {
  const float* feat;
  const bhalf* frag[10];          // w1_0,w2_0,wsc_0, w1_1,w2_1,wsc_1, w1_2,w2_2,wsc_2, final
  const float* g1[3]; const float* b1[3];
  const float* g2[3]; const float* b2[3];
  const float* gsc[3]; const float* bsc[3];
  const float* lin_b;
  float* out;
};

union AFrag { v16bf v; unsigned int u[8]; };
union BFrag { v16bf v; uint4 q[2]; };

// Gather one 16x32 bf16 A-fragment from an LDS row (or the shared Agg vector).
// Per ISA layout: lane<16 holds K={kb..kb+7, kb+16..kb+23}, lane>=16 the +8 set,
// packed as (even,odd) K pairs -> 8 dword LDS reads per lane.
__device__ __forceinline__ v16bf load_a(const bhalf* src, int kbase, int koff) {
  const unsigned int* p = (const unsigned int*)src;
  AFrag f;
  int b0 = (kbase + koff) >> 1;
#pragma unroll
  for (int i = 0; i < 4; ++i) f.u[i] = p[b0 + i];          // K = kb+koff+2i,+2i+1
#pragma unroll
  for (int i = 0; i < 4; ++i) f.u[4 + i] = p[b0 + 8 + i];  // K = kb+koff+16+2i,...
  return f.v;
}

__device__ __forceinline__ void zero4(v8f acc[4]) {
#pragma unroll
  for (int i = 0; i < 4; ++i)
#pragma unroll
    for (int r = 0; r < 8; ++r) acc[i][r] = 0.0f;
}

// One wave computes a 16-row M-strip x 64 cols. K<64 comes from per-row LDS
// (stride 72 halves); K>=64 comes from the shared Agg vector (the concat'ed
// segment_max half, identical for all rows of the cluster).
__device__ __forceinline__ void do_gemm(const bhalf* Alo, const bhalf* Ahi,
                                        const bhalf* __restrict__ frag,
                                        int ksteps, int wave, int lane,
                                        v8f acc[4]) {
  const int l16  = lane & 15;
  const int koff = (lane >> 4) << 3;
  const bhalf* xrow = Alo + (wave * 16 + l16) * 72;
  for (int ks = 0; ks < ksteps; ++ks) {
    v16bf a = (ks < 2) ? load_a(xrow, ks * 32, koff)
                       : load_a(Ahi, (ks - 2) * 32, koff);
    const bhalf* fb = frag + ks * 2048;
#pragma unroll
    for (int nt = 0; nt < 4; ++nt) {
      BFrag b;
      const uint4* q = (const uint4*)(fb + (nt * 32 + lane) * 16);
      b.q[0] = q[0];
      b.q[1] = q[1];
      acc[nt] = __builtin_amdgcn_wmma_f32_16x16x32_bf16(
          false, a, false, b.v, (short)0, acc[nt], false, false);
    }
  }
}

// Scatter fp32 accumulators to LDS per the C/D layout: lane<16 -> N=lane,
// rows 0..7; lane>=16 -> N=lane-16, rows 8..15 (per 16x16 tile).
__device__ __forceinline__ void store_acc(float* T, int wave, int lane,
                                          const v8f acc[4]) {
  const int col = lane & 15;
  const int rb  = wave * 16 + ((lane >> 4) << 3);
#pragma unroll
  for (int nt = 0; nt < 4; ++nt)
#pragma unroll
    for (int r = 0; r < 8; ++r)
      T[(rb + r) * 68 + nt * 16 + col] = acc[nt][r];
}

__device__ __forceinline__ void row_stats(const float* tr, float& mu, float& inv) {
  float s = 0.0f, s2 = 0.0f;
  for (int c = 0; c < 64; ++c) { float v = tr[c]; s += v; s2 += v * v; }
  mu = s * (1.0f / 64.0f);
  float var = s2 * (1.0f / 64.0f) - mu * mu;
  inv = rsqrtf(var + 1e-5f);
}

__device__ __forceinline__ void gn_relu_row(const float* tr, const float* g,
                                            const float* b, bhalf* out) {
  float mu, inv;
  row_stats(tr, mu, inv);
  for (int c = 0; c < 64; ++c) {
    float o = g[c] * (tr[c] - mu) * inv + b[c];
    out[c] = (bhalf)fmaxf(o, 0.0f);
  }
}

__device__ __forceinline__ void combine_row(const float* t1, const float* t2,
                                            const float* g2, const float* b2,
                                            const float* gs, const float* bs,
                                            bhalf* out) {
  float mu1, inv1, mu2, inv2;
  row_stats(t1, mu1, inv1);
  row_stats(t2, mu2, inv2);
  for (int c = 0; c < 64; ++c) {
    float o = g2[c] * (t1[c] - mu1) * inv1 + b2[c] +
              gs[c] * (t2[c] - mu2) * inv2 + bs[c];
    out[c] = (bhalf)fmaxf(o, 0.0f);
  }
}

// ROWS = points per cluster (64 lane / 32 veh), block = ROWS*2 threads
// (one wave32 per 16-row strip). Entire branch runs cluster-local in LDS.
template <int ROWS, int IN_DIM>
__global__ __launch_bounds__(ROWS * 2)
void branch_kernel(BranchArgs A, int cshift, int posoff, float flag0, float flag1) {
  __shared__ __align__(16) bhalf Xs[ROWS * 72];  // layer input, K<64 half (bf16)
  __shared__ __align__(16) bhalf Hs[ROWS * 72];  // relu(GN(x@w1)) (bf16)
  __shared__ __align__(16) bhalf Agg[72];        // segment_max, K>=64 half
  __shared__ float T1[ROWS * 68];
  __shared__ float T2[ROWS * 68];

  const int tid = threadIdx.x;
  const int wave = tid >> 5, lane = tid & 31;
  const int cluster = blockIdx.x;

  // Stage this cluster's raw features -> bf16, zero-padded to K=32.
  const float* f = A.feat + (size_t)cluster * ROWS * IN_DIM;
  for (int i = tid; i < ROWS * 32; i += ROWS * 2) {
    int r = i >> 5, k = i & 31;
    float v = (k < IN_DIM) ? f[r * IN_DIM + k] : 0.0f;
    Xs[r * 72 + k] = (bhalf)v;
  }
  __syncthreads();

#pragma unroll
  for (int l = 0; l < 3; ++l) {
    const int ksxw = (l == 0) ? 1 : 4;  // K-steps for x-side GEMMs

    v8f acc[4];
    zero4(acc);                                     // h1 = X @ W1
    do_gemm(Xs, Agg, A.frag[3 * l + 0], ksxw, wave, lane, acc);
    store_acc(T1, wave, lane, acc);
    __syncthreads();
    if (tid < ROWS) gn_relu_row(T1 + tid * 68, A.g1[l], A.b1[l], Hs + tid * 72);
    __syncthreads();

    zero4(acc);                                     // h2 = H @ W2 (K=64)
    do_gemm(Hs, Agg, A.frag[3 * l + 1], 2, wave, lane, acc);
    store_acc(T1, wave, lane, acc);

    zero4(acc);                                     // sc = X @ Wsc
    do_gemm(Xs, Agg, A.frag[3 * l + 2], ksxw, wave, lane, acc);
    store_acc(T2, wave, lane, acc);
    __syncthreads();

    // relu(GN(h2) + GN(sc)) -> new X (bf16)
    if (tid < ROWS)
      combine_row(T1 + tid * 68, T2 + tid * 68, A.g2[l], A.b2[l], A.gsc[l],
                  A.bsc[l], Xs + tid * 72);
    __syncthreads();

    // segment_max over the cluster -> Agg (serves as K in [64,128) next layer)
    if (tid < 64) {
      float m = -3.0e38f;
      for (int r = 0; r < ROWS; ++r) m = fmaxf(m, (float)Xs[r * 72 + tid]);
      Agg[tid] = (bhalf)m;
    }
    __syncthreads();
  }

  // final: [X|Agg] @ lin_w, + bias, segment_max -> one 64-vector per cluster
  v8f acc[4];
  zero4(acc);
  do_gemm(Xs, Agg, A.frag[9], 4, wave, lane, acc);
  store_acc(T1, wave, lane, acc);
  __syncthreads();

  float* orow =
      A.out +
      (size_t)(((cluster >> cshift) * 384) + posoff + (cluster & ((1 << cshift) - 1))) * 66;
  if (tid < 64) {
    float m = -3.0e38f;
    for (int r = 0; r < ROWS; ++r) m = fmaxf(m, T1[r * 68 + tid]);
    orow[tid] = m + A.lin_b[tid];  // max(x+b) == max(x)+b (b const per channel)
  }
  if (tid == 0) { orow[64] = flag0; orow[65] = flag1; }
}

// ---------------------------------------------------------------------------
// Host launch
// ---------------------------------------------------------------------------
extern "C" void kernel_launch(void* const* d_in, const int* in_sizes, int n_in,
                              void* d_out, int out_size, void* d_ws, size_t ws_size,
                              hipStream_t stream) {
  (void)in_sizes; (void)n_in; (void)out_size; (void)ws_size;
  auto P = [&](int i) { return (const float*)d_in[i]; };

  // GEMM table: w1_l, w2_l, wsc_l for l=0..2, then final lin_w
  static const int ksteps[10] = {1, 2, 1, 4, 2, 4, 4, 2, 4, 4};
  static const int laneWi[10] = {2, 5, 8, 11, 14, 17, 20, 23, 26, 56};
  static const int vehWi[10]  = {29, 32, 35, 38, 41, 44, 47, 50, 53, 58};
  static const int laneKs[10] = {8, 64, 8, 128, 64, 128, 128, 64, 128, 128};
  static const int vehKs[10]  = {10, 64, 10, 128, 64, 128, 128, 64, 128, 128};

  bhalf* ws = (bhalf*)d_ws;
  int laneOff[10], vehOff[10], o = 0;
  for (int g = 0; g < 10; ++g) { laneOff[g] = o; o += ksteps[g] * 2048; }
  for (int g = 0; g < 10; ++g) { vehOff[g]  = o; o += ksteps[g] * 2048; }

  for (int g = 0; g < 10; ++g) {
    int total = ksteps[g] * 2048;
    int blocks = (total + 255) / 256;
    pack_frag<<<blocks, 256, 0, stream>>>(P(laneWi[g]), ws + laneOff[g], laneKs[g], total);
    pack_frag<<<blocks, 256, 0, stream>>>(P(vehWi[g]),  ws + vehOff[g],  vehKs[g],  total);
  }

  // GN param indices (w1,g1,b1,w2,g2,b2,wsc,gsc,bsc per layer, insertion order)
  static const int lg[3][6] = {{3, 4, 6, 7, 9, 10},
                               {12, 13, 15, 16, 18, 19},
                               {21, 22, 24, 25, 27, 28}};
  static const int vg[3][6] = {{30, 31, 33, 34, 36, 37},
                               {39, 40, 42, 43, 45, 46},
                               {48, 49, 51, 52, 54, 55}};

  BranchArgs la = {}, va = {};
  la.feat = P(0);  la.lin_b = P(57); la.out = (float*)d_out;
  va.feat = P(1);  va.lin_b = P(59); va.out = (float*)d_out;
  for (int g = 0; g < 10; ++g) { la.frag[g] = ws + laneOff[g]; va.frag[g] = ws + vehOff[g]; }
  for (int l = 0; l < 3; ++l) {
    la.g1[l] = P(lg[l][0]); la.b1[l] = P(lg[l][1]);
    la.g2[l] = P(lg[l][2]); la.b2[l] = P(lg[l][3]);
    la.gsc[l] = P(lg[l][4]); la.bsc[l] = P(lg[l][5]);
    va.g1[l] = P(vg[l][0]); va.b1[l] = P(vg[l][1]);
    va.g2[l] = P(vg[l][2]); va.b2[l] = P(vg[l][3]);
    va.gsc[l] = P(vg[l][4]); va.bsc[l] = P(vg[l][5]);
  }

  // veh clusters occupy rows [0,128), lane rows [128,384) of each scene
  branch_kernel<64, 8><<<8192, 128, 0, stream>>>(la, 8, 128, 0.0f, 1.0f);
  branch_kernel<32, 10><<<4096, 64, 0, stream>>>(va, 7, 0, 1.0f, 0.0f);
}
